// PointNetFeaturePropagation_91328184582664
// MI455X (gfx1250) — matrix operand order repaired and checked by reading, not compiled, not run on hardware
//
#include <hip/hip_runtime.h>
#include <hip/hip_bf16.h>

typedef __attribute__((ext_vector_type(16))) __bf16 v16bf;
typedef __attribute__((ext_vector_type(8)))  __bf16 v8bf;
typedef __attribute__((ext_vector_type(4)))  __bf16 v4bf;
typedef __attribute__((ext_vector_type(8)))  float  v8f;
typedef __attribute__((ext_vector_type(4)))  float  v4f;

#define BB   8
#define NN   8192
#define SS   2048
#define D1C  128
#define D2C  256
#define INCH 384
#define C0C  256
#define C1C  128
#define MM   (BB * NN)   // 65536 rows

// ---------------------------------------------------------------------------
// prep: convert weights f32 -> bf16, zero BN stat accumulators
// ---------------------------------------------------------------------------
__global__ __launch_bounds__(256) void prep_kernel(const float* __restrict__ W0,
                                                   const float* __restrict__ W1,
                                                   __bf16* __restrict__ w0b,
                                                   __bf16* __restrict__ w1b,
                                                   float* __restrict__ stats) {
    const int n0 = C0C * INCH;          // 98304
    const int n1 = C1C * C0C;           // 32768
    const int ns = 2 * C0C + 2 * C1C;   // 768
    int idx = blockIdx.x * 256 + threadIdx.x;
    if (idx < n0)                 w0b[idx] = (__bf16)W0[idx];
    else if (idx < n0 + n1)       w1b[idx - n0] = (__bf16)W1[idx - n0];
    else if (idx < n0 + n1 + ns)  stats[idx - n0 - n1] = 0.0f;
}

// ---------------------------------------------------------------------------
// transpose points2 [B,D2,S] -> p2t [B,S,D2]  (coalesced via LDS 32x32 tiles)
// ---------------------------------------------------------------------------
__global__ __launch_bounds__(256) void transpose_p2_kernel(const float* __restrict__ p2,
                                                           float* __restrict__ p2t) {
    __shared__ float tile[32][33];
    int s0 = blockIdx.x * 32, c0 = blockIdx.y * 32, b = blockIdx.z;
    const float* src = p2  + (size_t)b * D2C * SS;
    float*       dst = p2t + (size_t)b * SS * D2C;
    for (int i = threadIdx.x; i < 1024; i += 256) {
        int cc = i >> 5, ss = i & 31;
        tile[cc][ss] = src[(size_t)(c0 + cc) * SS + s0 + ss];
    }
    __syncthreads();
    for (int i = threadIdx.x; i < 1024; i += 256) {
        int ss = i >> 5, cc = i & 31;
        dst[(size_t)(s0 + ss) * D2C + c0 + cc] = tile[cc][ss];
    }
}

// ---------------------------------------------------------------------------
// points1 [B,D1,N] -> X[:, 0:128] bf16 (row = point), LDS tile transpose
// ---------------------------------------------------------------------------
__global__ __launch_bounds__(256) void copy_p1_kernel(const float* __restrict__ p1,
                                                      __bf16* __restrict__ X) {
    __shared__ float tile[32][33];
    int n0 = blockIdx.x * 32, c0 = blockIdx.y * 32, b = blockIdx.z;
    const float* src = p1 + (size_t)b * D1C * NN;
    __bf16*      dst = X  + (size_t)b * NN * INCH;
    for (int i = threadIdx.x; i < 1024; i += 256) {
        int cc = i >> 5, nn = i & 31;
        tile[cc][nn] = src[(size_t)(c0 + cc) * NN + n0 + nn];
    }
    __syncthreads();
    for (int i = threadIdx.x; i < 1024; i += 256) {
        int nn = i >> 5, cc = i & 31;
        dst[(size_t)(n0 + nn) * INCH + c0 + cc] = (__bf16)tile[cc][nn];
    }
}

// ---------------------------------------------------------------------------
// 3-NN interpolation: xyz2 of one batch resident in LDS (24KB of 320KB WGP LDS)
// phase 1: per-thread scan of 2048 candidates keeping 3 smallest
// phase 2: cooperative gather, thread == channel -> fully coalesced row reads
// writes bf16 into X[:, 128:384]
// ---------------------------------------------------------------------------
__global__ __launch_bounds__(256) void interp_kernel(const float* __restrict__ xyz1,
                                                     const float* __restrict__ xyz2,
                                                     const float* __restrict__ p2t,
                                                     __bf16* __restrict__ X) {
    __shared__ float sx[SS], sy[SS], sz[SS];
    __shared__ int   sidx[256][3];
    __shared__ float swt[256][3];
    int b = blockIdx.x >> 5;       // 32 chunks of 256 points per batch
    int chunk = blockIdx.x & 31;
    int tid = threadIdx.x;

    const float* x2 = xyz2 + (size_t)b * 3 * SS;
    for (int i = tid; i < SS; i += 256) {
        sx[i] = x2[i];
        sy[i] = x2[SS + i];
        sz[i] = x2[2 * SS + i];
    }
    __syncthreads();

    int n = chunk * 256 + tid;
    const float* x1 = xyz1 + (size_t)b * 3 * NN;
    float px = x1[n], py = x1[NN + n], pz = x1[2 * NN + n];

    float d0 = 3.4e38f, d1 = 3.4e38f, d2 = 3.4e38f;
    int   i0 = 0, i1 = 0, i2 = 0;
    for (int s = 0; s < SS; ++s) {
        float dx = px - sx[s], dy = py - sy[s], dz = pz - sz[s];
        float d = dx * dx + dy * dy + dz * dz;
        if (d < d0)      { d2 = d1; i2 = i1; d1 = d0; i1 = i0; d0 = d; i0 = s; }
        else if (d < d1) { d2 = d1; i2 = i1; d1 = d;  i1 = s; }
        else if (d < d2) { d2 = d;  i2 = s; }
    }
    float r0 = 1.0f / (d0 + 1e-8f);
    float r1 = 1.0f / (d1 + 1e-8f);
    float r2 = 1.0f / (d2 + 1e-8f);
    float rs = 1.0f / (r0 + r1 + r2);
    sidx[tid][0] = i0; sidx[tid][1] = i1; sidx[tid][2] = i2;
    swt[tid][0] = r0 * rs; swt[tid][1] = r1 * rs; swt[tid][2] = r2 * rs;
    __syncthreads();

    const float* pb = p2t + (size_t)b * SS * D2C;
    for (int p = 0; p < 256; ++p) {
        int   j0 = sidx[p][0], j1 = sidx[p][1], j2 = sidx[p][2];
        float w0 = swt[p][0],  w1 = swt[p][1],  w2 = swt[p][2];
        float v = w0 * pb[(size_t)j0 * D2C + tid]
                + w1 * pb[(size_t)j1 * D2C + tid]
                + w2 * pb[(size_t)j2 * D2C + tid];
        size_t m = (size_t)b * NN + (size_t)chunk * 256 + p;
        X[m * INCH + D1C + tid] = (__bf16)v;
    }
}

// ---------------------------------------------------------------------------
// WMMA bf16 GEMM:  Y[M, NTOT] = A[M, K] * W[NTOT, K]^T + bias
// wave32, each wave produces a 32x32 f32 tile with 4 v_wmma_f32_16x16x32_bf16
// per 32-wide K step. Fragment layouts per CDNA5 ISA.
// ---------------------------------------------------------------------------
__device__ __forceinline__ v16bf load_a_frag(const __bf16* p) {
    v8bf lo = *(const v8bf*)p;          // K = base .. base+7   (this half)
    v8bf hi = *(const v8bf*)(p + 16);   // K = base+16 .. +23   (this half)
    v16bf a;
#pragma unroll
    for (int i = 0; i < 8; ++i) { a[i] = lo[i]; a[i + 8] = hi[i]; }
    return a;
}

template <int NTOT, int K>
__global__ __launch_bounds__(256) void gemm_bf16_kernel(const __bf16* __restrict__ A,
                                                        const __bf16* __restrict__ W,
                                                        const float* __restrict__ bias,
                                                        float* __restrict__ Y) {
    constexpr int TILES_N = NTOT / 32;
    int wave = (blockIdx.x * 256 + threadIdx.x) >> 5;
    int lane = threadIdx.x & 31;
    int tn = wave % TILES_N;
    int tm = wave / TILES_N;
    int half = lane >> 4;
    int r    = lane & 15;
    size_t m0 = (size_t)tm * 32;
    int    n0 = tn * 32;

    // A: lane r holds row m0+r (+16), K elements {h*8..h*8+7, 16+h*8..16+h*8+7}
    const __bf16* a0p = A + (m0 + r) * (size_t)K + half * 8;
    const __bf16* a1p = a0p + (size_t)16 * K;
    // B: lane r holds column n0+r (+16), K elements {h*8..h*8+7, 16+h*8..16+h*8+7}
    const __bf16* b0p = W + (size_t)(n0 + r) * K + half * 8;
    const __bf16* b1p = b0p + (size_t)16 * K;

    v8f acc00 = {}, acc01 = {}, acc10 = {}, acc11 = {};
#pragma unroll 2
    for (int k0 = 0; k0 < K; k0 += 32) {
        v16bf a0 = load_a_frag(a0p + k0);
        v16bf a1 = load_a_frag(a1p + k0);
        v16bf b0 = load_a_frag(b0p + k0);
        v16bf b1 = load_a_frag(b1p + k0);
        acc00 = __builtin_amdgcn_wmma_f32_16x16x32_bf16(false, a0, false, b0, (short)0, acc00, false, false);
        acc01 = __builtin_amdgcn_wmma_f32_16x16x32_bf16(false, a0, false, b1, (short)0, acc01, false, false);
        acc10 = __builtin_amdgcn_wmma_f32_16x16x32_bf16(false, a1, false, b0, (short)0, acc10, false, false);
        acc11 = __builtin_amdgcn_wmma_f32_16x16x32_bf16(false, a1, false, b1, (short)0, acc11, false, false);
    }

    // C/D layout: lane (0-15,16-31) -> N = n0+r ; VGPR rr -> M = m0 + half*8 + rr
    int    ncol = n0 + r;
    size_t mrow = m0 + half * 8;
    float bias0 = bias[ncol];
    float bias1 = bias[ncol + 16];
#pragma unroll
    for (int rr = 0; rr < 8; ++rr) {
        size_t row0 = (mrow + rr) * (size_t)NTOT;
        size_t row1 = (mrow + 16 + rr) * (size_t)NTOT;
        Y[row0 + ncol]      = acc00[rr] + bias0;
        Y[row0 + ncol + 16] = acc01[rr] + bias1;
        Y[row1 + ncol]      = acc10[rr] + bias0;
        Y[row1 + ncol + 16] = acc11[rr] + bias1;
    }
}

// ---------------------------------------------------------------------------
// BN statistics: per-channel sum / sumsq over all M rows (coalesced reads)
// ---------------------------------------------------------------------------
template <int C>
__global__ __launch_bounds__(256) void stats_kernel(const float* __restrict__ Y,
                                                    float* __restrict__ sum,
                                                    float* __restrict__ sumsq) {
    constexpr int RPT  = 256 / C;   // rows covered per iteration
    constexpr int ROWS = 256;       // rows per block
    int tid = threadIdx.x;
    int c   = tid % C;
    int row = blockIdx.x * ROWS + tid / C;
    int rowEnd = blockIdx.x * ROWS + ROWS;
    float s = 0.0f, q = 0.0f;
    for (; row < rowEnd; row += RPT) {
        float v = Y[(size_t)row * C + c];
        s += v;
        q += v * v;
    }
    __shared__ float ls[256], lq[256];
    ls[tid] = s; lq[tid] = q;
    __syncthreads();
    if (tid < C) {
        float ts = ls[tid], tq = lq[tid];
        for (int o = C; o < 256; o += C) { ts += ls[tid + o]; tq += lq[tid + o]; }
        atomicAdd(&sum[c], ts);
        atomicAdd(&sumsq[c], tq);
    }
}

// ---------------------------------------------------------------------------
// Fold BN into per-channel affine: y' = y*a + b ; a = g*rsqrt(var+eps), b = be - mean*a
// ---------------------------------------------------------------------------
__global__ void finalize_kernel(const float* __restrict__ sum, const float* __restrict__ sumsq,
                                const float* __restrict__ g, const float* __restrict__ be,
                                float* __restrict__ Ac, float* __restrict__ Bc, int C) {
    int c = blockIdx.x * blockDim.x + threadIdx.x;
    if (c >= C) return;
    float inv  = 1.0f / (float)MM;
    float mean = sum[c] * inv;
    float var  = sumsq[c] * inv - mean * mean;
    float a    = g[c] * rsqrtf(var + 1e-5f);
    Ac[c] = a;
    Bc[c] = be[c] - mean * a;
}

// ---------------------------------------------------------------------------
// BN + ReLU + f32 -> bf16 (layer-0 activation for GEMM1), vectorized x4
// ---------------------------------------------------------------------------
__global__ __launch_bounds__(256) void bnrelu_bf16_kernel(const float* __restrict__ Y,
                                                          const float* __restrict__ Ac,
                                                          const float* __restrict__ Bc,
                                                          __bf16* __restrict__ Xo) {
    size_t idx = ((size_t)blockIdx.x * 256 + threadIdx.x) * 4;
    v4f v = *(const v4f*)(Y + idx);
    int c = (int)(idx & (C0C - 1));
    v4bf o;
#pragma unroll
    for (int i = 0; i < 4; ++i) {
        float t = fmaxf(v[i] * Ac[c + i] + Bc[c + i], 0.0f);
        o[i] = (__bf16)t;
    }
    *(v4bf*)(Xo + idx) = o;
}

// ---------------------------------------------------------------------------
// Final: BN + ReLU on Y1 [M,128] and transpose to out [B,128,N] via LDS tiles
// ---------------------------------------------------------------------------
__global__ __launch_bounds__(256) void bnrelu_tr_out_kernel(const float* __restrict__ Y,
                                                            const float* __restrict__ Ac,
                                                            const float* __restrict__ Bc,
                                                            float* __restrict__ out) {
    __shared__ float tile[32][33];
    int n0 = blockIdx.x * 32, c0 = blockIdx.y * 32, b = blockIdx.z;
    const float* src = Y   + (size_t)b * NN * C1C;
    float*       dst = out + (size_t)b * C1C * NN;
    for (int i = threadIdx.x; i < 1024; i += 256) {
        int nn = i >> 5, cc = i & 31;
        int c = c0 + cc;
        float v = src[(size_t)(n0 + nn) * C1C + c];
        tile[nn][cc] = fmaxf(v * Ac[c] + Bc[c], 0.0f);
    }
    __syncthreads();
    for (int i = threadIdx.x; i < 1024; i += 256) {
        int cc = i >> 5, nn = i & 31;
        dst[(size_t)(c0 + cc) * NN + n0 + nn] = tile[nn][cc];
    }
}

// ---------------------------------------------------------------------------
// Host launch
// ---------------------------------------------------------------------------
extern "C" void kernel_launch(void* const* d_in, const int* in_sizes, int n_in,
                              void* d_out, int out_size, void* d_ws, size_t ws_size,
                              hipStream_t stream) {
    (void)in_sizes; (void)n_in; (void)out_size; (void)ws_size;

    const float* xyz1    = (const float*)d_in[0];
    const float* xyz2    = (const float*)d_in[1];
    const float* points1 = (const float*)d_in[2];
    const float* points2 = (const float*)d_in[3];
    const float* W0      = (const float*)d_in[4];
    const float* b0      = (const float*)d_in[5];
    const float* g0      = (const float*)d_in[6];
    const float* be0     = (const float*)d_in[7];
    const float* W1      = (const float*)d_in[8];
    const float* b1      = (const float*)d_in[9];
    const float* g1      = (const float*)d_in[10];
    const float* be1     = (const float*)d_in[11];
    float* out = (float*)d_out;

    // workspace layout (bytes), all offsets >= 64B aligned
    constexpr size_t OFF_X    = 0;                                       // bf16 [M,384]
    constexpr size_t OFF_P2T  = OFF_X    + (size_t)MM * INCH * 2;        // f32  [B,S,256]
    constexpr size_t OFF_Y0   = OFF_P2T  + (size_t)BB * SS * D2C * 4;    // f32  [M,256]
    constexpr size_t OFF_X1   = OFF_Y0   + (size_t)MM * C0C * 4;         // bf16 [M,256]
    constexpr size_t OFF_Y1   = OFF_X1   + (size_t)MM * C0C * 2;         // f32  [M,128]
    constexpr size_t OFF_W0B  = OFF_Y1   + (size_t)MM * C1C * 4;         // bf16 [256,384]
    constexpr size_t OFF_W1B  = OFF_W0B  + (size_t)C0C * INCH * 2;       // bf16 [128,256]
    constexpr size_t OFF_STAT = OFF_W1B  + (size_t)C1C * C0C * 2;        // f32  [768]
    constexpr size_t OFF_COEF = OFF_STAT + 768 * 4;                      // f32  [768]

    char* ws = (char*)d_ws;
    __bf16* X    = (__bf16*)(ws + OFF_X);
    float*  p2t  = (float*) (ws + OFF_P2T);
    float*  Y0   = (float*) (ws + OFF_Y0);
    __bf16* X1   = (__bf16*)(ws + OFF_X1);
    float*  Y1   = (float*) (ws + OFF_Y1);
    __bf16* w0b  = (__bf16*)(ws + OFF_W0B);
    __bf16* w1b  = (__bf16*)(ws + OFF_W1B);
    float*  stat = (float*) (ws + OFF_STAT);
    float*  coef = (float*) (ws + OFF_COEF);

    float* sum0 = stat;        float* sq0 = stat + 256;
    float* sum1 = stat + 512;  float* sq1 = stat + 640;
    float* A0 = coef;          float* B0 = coef + 256;
    float* A1 = coef + 512;    float* B1 = coef + 640;

    // 1) weight conversion + stats zeroing
    prep_kernel<<<(C0C * INCH + C1C * C0C + 768 + 255) / 256, 256, 0, stream>>>(W0, W1, w0b, w1b, stat);
    // 2) points2 transpose, points1 -> X[:,0:128]
    transpose_p2_kernel<<<dim3(SS / 32, D2C / 32, BB), 256, 0, stream>>>(points2, p2t);
    copy_p1_kernel<<<dim3(NN / 32, D1C / 32, BB), 256, 0, stream>>>(points1, X);
    // 3) 3-NN interpolation -> X[:,128:384]
    interp_kernel<<<BB * (NN / 256), 256, 0, stream>>>(xyz1, xyz2, p2t, X);
    // 4) layer 0: GEMM + bias, BN stats, fold, BN+ReLU -> bf16
    gemm_bf16_kernel<C0C, INCH><<<(MM / 32) * (C0C / 32) / 8, 256, 0, stream>>>(X, w0b, b0, Y0);
    stats_kernel<C0C><<<MM / 256, 256, 0, stream>>>(Y0, sum0, sq0);
    finalize_kernel<<<1, 256, 0, stream>>>(sum0, sq0, g0, be0, A0, B0, C0C);
    bnrelu_bf16_kernel<<<(MM * C0C / 4) / 256, 256, 0, stream>>>(Y0, A0, B0, X1);
    // 5) layer 1: GEMM + bias, BN stats, fold, BN+ReLU + transpose to [B,128,N]
    gemm_bf16_kernel<C1C, C0C><<<(MM / 32) * (C1C / 32) / 8, 256, 0, stream>>>(X1, w1b, b1, Y1);
    stats_kernel<C1C><<<MM / 256, 256, 0, stream>>>(Y1, sum1, sq1);
    finalize_kernel<<<1, 128, 0, stream>>>(sum1, sq1, g1, be1, A1, B1, C1C);
    bnrelu_tr_out_kernel<<<dim3(NN / 32, C1C / 32, BB), 256, 0, stream>>>(Y1, A1, B1, out);
}
